// PDEPredictor_27779848471105
// MI455X (gfx1250) — compile-verified, matrix operand order checked by reading, not run on hardware
//
#include <hip/hip_runtime.h>
#include <hip/hip_bf16.h>
#include <math.h>

typedef __bf16 bf16;
typedef __attribute__((ext_vector_type(16))) __bf16 v16bf;
typedef __attribute__((ext_vector_type(8)))  float  v8f;

constexpr int kB   = 8;
constexpr int kT   = 8;
constexpr int kC   = 128;
constexpr int kH   = 64;
constexpr int kW   = 64;
constexpr int kA   = 32;
constexpr int kHID = 256;
constexpr int kHW  = kH * kW;          // 4096
constexpr int kM   = kB * kHW;         // 32768 pixels
constexpr float kEPS = 1e-6f;

// ---------- small device helpers ----------
static __device__ __forceinline__ float dev_sigmoid(float x) { return 1.0f / (1.0f + __expf(-x)); }
static __device__ __forceinline__ float dev_softplus(float x) { return (x > 20.0f) ? x : log1pf(__expf(x)); }
static __device__ __forceinline__ float dev_gelu(float x) { return 0.5f * x * (1.0f + erff(x * 0.70710678118654752f)); }

static __device__ __forceinline__ void vzero(v8f& v) {
    #pragma unroll
    for (int j = 0; j < 8; ++j) v[j] = 0.0f;
}

// Cooperative stage of a 128-row x 32-K bf16 tile into LDS.
// src must point at (row0, k0); ld = row leading dim in elements. 256 threads.
static __device__ __forceinline__ void stage_tile(bf16* dst, const bf16* __restrict__ src,
                                                  int ld, int tid) {
    const int row = tid >> 1, half = tid & 1;
    *(v16bf*)(&dst[row * 32 + half * 16]) =
        *(const v16bf*)(&src[(size_t)row * ld + half * 16]);
}

// Fragment (16 rows x 32 K) from a staged LDS tile; r16 selects which 16-row group.
static __device__ __forceinline__ v16bf frag_lds(const bf16* t, int r16, int lane) {
    return *(const v16bf*)(&t[(r16 * 16 + (lane & 15)) * 32 + (lane >> 4) * 16]);
}

static __device__ __forceinline__ void add4(float4& s, const float* p) {
    const float4 v = *(const float4*)p;
    s.x += v.x; s.y += v.y; s.z += v.z; s.w += v.w;
}

// ---------- one-time prep kernels ----------
__global__ void k_scal(const float* __restrict__ log_dt, const float* __restrict__ state_decay,
                       const float* __restrict__ log_diff, float* __restrict__ scal,
                       float* __restrict__ strengths) {
    const int tid = threadIdx.x;
    if (tid == 0) {
        scal[0] = dev_softplus(log_dt[0]);                    // dt
        scal[1] = 0.5f + 0.49f * dev_sigmoid(state_decay[0]); // decay
    }
    if (tid < 3 * kC) strengths[tid] = dev_softplus(log_diff[tid]);
}

__global__ void k_acts(const float* __restrict__ cin, const float* __restrict__ ap_w,
                       const float* __restrict__ ap_b, float* __restrict__ acts) {
    const int idx = blockIdx.x * blockDim.x + threadIdx.x;
    if (idx >= kB * kT * kC) return;
    const int cc = idx % kC;
    const int bt = idx / kC;
    float s = ap_b[cc];
    #pragma unroll 4
    for (int a = 0; a < kA; ++a) s += cin[bt * kA + a] * ap_w[cc * kA + a];
    acts[idx] = s;
}

__global__ void k_cvt(const float* __restrict__ src, bf16* __restrict__ dst, int n) {
    const int i = blockIdx.x * blockDim.x + threadIdx.x;
    if (i < n) dst[i] = (bf16)src[i];
}

// NCHW fp32 observation slice -> NHWC bf16 (LDS-tiled transpose)
__global__ __launch_bounds__(256) void k_transpose_obs(const float* __restrict__ x,
                                                       bf16* __restrict__ obs, int t) {
    __shared__ float tile[32][33];
    const int tx = threadIdx.x & 31;
    const int ty = threadIdx.x >> 5;              // 0..7
    const int p0 = blockIdx.x * 32;               // pixel base inside image
    const int c0 = blockIdx.y * 32;               // channel base
    const int b  = blockIdx.z;
    const float* src = x + ((size_t)(b * kT + t) * kC) * kHW;
    #pragma unroll
    for (int i = 0; i < 4; ++i) {
        const int cc = c0 + ty + i * 8;
        tile[ty + i * 8][tx] = src[(size_t)cc * kHW + p0 + tx];
    }
    __syncthreads();
    #pragma unroll
    for (int i = 0; i < 4; ++i) {
        const int p = p0 + ty + i * 8;
        obs[(size_t)(b * kHW + p) * kC + c0 + tx] = (bf16)tile[tx][ty + i * 8];
    }
}

// ---------- fused gate/value WMMA GEMM: state = decay*state + sigmoid(Wg x)·tanh(Wv x) ----------
__global__ __launch_bounds__(256) void k_gate_value(const bf16* __restrict__ obs,
                                                    const bf16* __restrict__ Wg,
                                                    const bf16* __restrict__ Wv,
                                                    const float* __restrict__ bg,
                                                    const float* __restrict__ bv,
                                                    const float* __restrict__ scal,
                                                    float* __restrict__ state,
                                                    bf16* __restrict__ state_bf) {
    __shared__ bf16 Atile[128 * 32];
    __shared__ bf16 Bgt[128 * 32];
    __shared__ bf16 Bvt[128 * 32];
    const int tid = threadIdx.x, wave = tid >> 5, lane = tid & 31;
    const int m0 = blockIdx.x * 128;
    const float decay = scal[1];
    v8f accg[8], accv[8];
    #pragma unroll
    for (int i = 0; i < 8; ++i) { vzero(accg[i]); vzero(accv[i]); }

    for (int k0 = 0; k0 < kC; k0 += 32) {
        __syncthreads();
        stage_tile(Atile, obs + (size_t)m0 * kC + k0, kC, tid);
        stage_tile(Bgt, Wg + k0, kC, tid);
        stage_tile(Bvt, Wv + k0, kC, tid);
        __syncthreads();
        if (k0 + 32 < kC) {   // overlap: prefetch next A k-chunk while MACs run
            __builtin_prefetch(obs + (size_t)(m0 + (tid >> 1)) * kC + k0 + 32 + (tid & 1) * 16, 0, 1);
        }
        const v16bf a = frag_lds(Atile, wave, lane);
        #pragma unroll
        for (int nt = 0; nt < 8; ++nt) {
            const v16bf bgf = frag_lds(Bgt, nt, lane);
            const v16bf bvf = frag_lds(Bvt, nt, lane);
            accg[nt] = __builtin_amdgcn_wmma_f32_16x16x32_bf16(false, a, false, bgf, (short)0, accg[nt], false, false);
            accv[nt] = __builtin_amdgcn_wmma_f32_16x16x32_bf16(false, a, false, bvf, (short)0, accv[nt], false, false);
        }
    }
    const int mbase = m0 + wave * 16 + ((lane >> 4) << 3);
    const int nl = lane & 15;
    #pragma unroll
    for (int nt = 0; nt < 8; ++nt) {
        const int n = nt * 16 + nl;
        const float gb = bg[n], vb = bv[n];
        #pragma unroll
        for (int r = 0; r < 8; ++r) {
            const size_t idx = (size_t)(mbase + r) * kC + n;
            const float g = dev_sigmoid(accg[nt][r] + gb);
            const float v = tanhf(accv[nt][r] + vb);
            const float s = decay * state[idx] + g * v;
            state[idx] = s;
            state_bf[idx] = (bf16)s;
        }
    }
}

// ---------- reaction stage 1: hidden = gelu(r1_w @ state + r1_b), output bf16 ----------
__global__ __launch_bounds__(256) void k_gemm_r1(const bf16* __restrict__ Sbf,
                                                 const bf16* __restrict__ W1,
                                                 const float* __restrict__ b1,
                                                 bf16* __restrict__ hidden) {
    __shared__ bf16 Atile[128 * 32];
    __shared__ bf16 Btile[128 * 32];
    const int tid = threadIdx.x, wave = tid >> 5, lane = tid & 31;
    const int m0 = blockIdx.x * 128;
    const int n0 = blockIdx.y * 128;
    v8f acc[8];
    #pragma unroll
    for (int i = 0; i < 8; ++i) vzero(acc[i]);

    for (int k0 = 0; k0 < kC; k0 += 32) {
        __syncthreads();
        stage_tile(Atile, Sbf + (size_t)m0 * kC + k0, kC, tid);
        stage_tile(Btile, W1 + (size_t)n0 * kC + k0, kC, tid);
        __syncthreads();
        if (k0 + 32 < kC) {
            __builtin_prefetch(Sbf + (size_t)(m0 + (tid >> 1)) * kC + k0 + 32 + (tid & 1) * 16, 0, 1);
        }
        const v16bf a = frag_lds(Atile, wave, lane);
        #pragma unroll
        for (int nt = 0; nt < 8; ++nt) {
            const v16bf b = frag_lds(Btile, nt, lane);
            acc[nt] = __builtin_amdgcn_wmma_f32_16x16x32_bf16(false, a, false, b, (short)0, acc[nt], false, false);
        }
    }
    const int mbase = m0 + wave * 16 + ((lane >> 4) << 3);
    const int nl = lane & 15;
    #pragma unroll
    for (int nt = 0; nt < 8; ++nt) {
        const int n = n0 + nt * 16 + nl;
        const float bsum = b1[n];
        #pragma unroll
        for (int r = 0; r < 8; ++r) {
            hidden[(size_t)(mbase + r) * kHID + n] = (bf16)dev_gelu(acc[nt][r] + bsum);
        }
    }
}

// ---------- diffusion: three dilated 5-point Laplacians, NHWC float4 ----------
__global__ __launch_bounds__(256) void k_diffusion(const float* __restrict__ S,
                                                   const float* __restrict__ strengths,
                                                   float* __restrict__ U) {
    const int idx = blockIdx.x * blockDim.x + threadIdx.x;
    if (idx >= kM * (kC / 4)) return;
    const int cg = idx & 31;          // 32 channel groups of 4
    const int p  = idx >> 5;          // pixel
    const int c  = cg * 4;
    const int b = p >> 12, hw = p & 4095, h = hw >> 6, w = hw & 63;
    const float4 ctr = *(const float4*)(S + (size_t)p * kC + c);
    float4 acc = {0.f, 0.f, 0.f, 0.f};
    #pragma unroll
    for (int i = 0; i < 3; ++i) {
        const int d = 1 << i;         // dilations 1, 2, 4
        float4 sum = {0.f, 0.f, 0.f, 0.f};
        if (h >= d)     add4(sum, S + ((size_t)((b << 12) + (h - d) * kW + w)) * kC + c);
        if (h + d < kH) add4(sum, S + ((size_t)((b << 12) + (h + d) * kW + w)) * kC + c);
        if (w >= d)     add4(sum, S + ((size_t)p - d) * kC + c);
        if (w + d < kW) add4(sum, S + ((size_t)p + d) * kC + c);
        const float4 st = *(const float4*)(strengths + i * kC + c);
        acc.x += st.x * (sum.x - 4.0f * ctr.x);
        acc.y += st.y * (sum.y - 4.0f * ctr.y);
        acc.z += st.z * (sum.z - 4.0f * ctr.z);
        acc.w += st.w * (sum.w - 4.0f * ctr.w);
    }
    *(float4*)(U + (size_t)p * kC + c) = acc;
}

// ---------- reaction stage 2 fused with Euler integration and (optionally) RMSNorm ----------
__global__ __launch_bounds__(256) void k_gemm_r2(const bf16* __restrict__ hidden,
                                                 const bf16* __restrict__ W2,
                                                 const float* __restrict__ b2,
                                                 const float* __restrict__ upd,
                                                 const float* __restrict__ acts,
                                                 const float* __restrict__ scal,
                                                 const float* __restrict__ norm_w,
                                                 int t, int doNorm,
                                                 float* __restrict__ state,
                                                 bf16* __restrict__ state_bf) {
    __shared__ bf16 Atile[128 * 32];
    __shared__ bf16 Btile[128 * 32];
    const int tid = threadIdx.x, wave = tid >> 5, lane = tid & 31;
    const int m0 = blockIdx.x * 128;
    const float dt = scal[0];
    v8f acc[8];
    #pragma unroll
    for (int i = 0; i < 8; ++i) vzero(acc[i]);

    for (int k0 = 0; k0 < kHID; k0 += 32) {
        __syncthreads();
        stage_tile(Atile, hidden + (size_t)m0 * kHID + k0, kHID, tid);
        stage_tile(Btile, W2 + k0, kHID, tid);
        __syncthreads();
        if (k0 + 32 < kHID) {
            __builtin_prefetch(hidden + (size_t)(m0 + (tid >> 1)) * kHID + k0 + 32 + (tid & 1) * 16, 0, 1);
        }
        const v16bf a = frag_lds(Atile, wave, lane);
        #pragma unroll
        for (int nt = 0; nt < 8; ++nt) {
            const v16bf b = frag_lds(Btile, nt, lane);
            acc[nt] = __builtin_amdgcn_wmma_f32_16x16x32_bf16(false, a, false, b, (short)0, acc[nt], false, false);
        }
    }
    const int mbase = m0 + wave * 16 + ((lane >> 4) << 3);
    const int nl = lane & 15;

    // integrate: acc <- state + dt*(r2 + b2 + diffusion + action)
    #pragma unroll
    for (int nt = 0; nt < 8; ++nt) {
        const int n = nt * 16 + nl;
        const float bsum = b2[n];
        #pragma unroll
        for (int r = 0; r < 8; ++r) {
            const int m = mbase + r;
            const size_t idx = (size_t)m * kC + n;
            const int bb = m >> 12;
            const float val = acc[nt][r] + bsum + upd[idx] + acts[((size_t)bb * kT + t) * kC + n];
            acc[nt][r] = state[idx] + dt * val;
        }
    }
    // optional fused RMSNorm over channels: this block owns all 128 channels of its rows.
    // Row m's channels live in one 16-lane half (n = nt*16 + lane&15) -> xor-reduce masks 1,2,4,8.
    if (doNorm) {
        #pragma unroll
        for (int r = 0; r < 8; ++r) {
            float ss = 0.0f;
            #pragma unroll
            for (int nt = 0; nt < 8; ++nt) ss += acc[nt][r] * acc[nt][r];
            #pragma unroll
            for (int mask = 1; mask < 16; mask <<= 1) ss += __shfl_xor(ss, mask, 32);
            const float sc = rsqrtf(ss * (1.0f / kC) + kEPS);
            #pragma unroll
            for (int nt = 0; nt < 8; ++nt) acc[nt][r] *= sc;
        }
    }
    #pragma unroll
    for (int nt = 0; nt < 8; ++nt) {
        const int n = nt * 16 + nl;
        const float nw = doNorm ? norm_w[n] : 1.0f;
        #pragma unroll
        for (int r = 0; r < 8; ++r) {
            const size_t idx = (size_t)(mbase + r) * kC + n;
            const float ns = acc[nt][r] * nw;
            state[idx] = ns;
            state_bf[idx] = (bf16)ns;
        }
    }
}

// ---------- final projection, writes NCHW fp32 output ----------
__global__ __launch_bounds__(256) void k_final(const bf16* __restrict__ Sbf,
                                               const bf16* __restrict__ Wo,
                                               const float* __restrict__ bo,
                                               float* __restrict__ out) {
    __shared__ bf16 Atile[128 * 32];
    __shared__ bf16 Btile[128 * 32];
    const int tid = threadIdx.x, wave = tid >> 5, lane = tid & 31;
    const int m0 = blockIdx.x * 128;
    v8f acc[8];
    #pragma unroll
    for (int i = 0; i < 8; ++i) vzero(acc[i]);

    for (int k0 = 0; k0 < kC; k0 += 32) {
        __syncthreads();
        stage_tile(Atile, Sbf + (size_t)m0 * kC + k0, kC, tid);
        stage_tile(Btile, Wo + k0, kC, tid);
        __syncthreads();
        const v16bf a = frag_lds(Atile, wave, lane);
        #pragma unroll
        for (int nt = 0; nt < 8; ++nt) {
            const v16bf b = frag_lds(Btile, nt, lane);
            acc[nt] = __builtin_amdgcn_wmma_f32_16x16x32_bf16(false, a, false, b, (short)0, acc[nt], false, false);
        }
    }
    const int mbase = m0 + wave * 16 + ((lane >> 4) << 3);
    const int nl = lane & 15;
    #pragma unroll
    for (int nt = 0; nt < 8; ++nt) {
        const int n = nt * 16 + nl;
        const float bsum = bo[n];
        #pragma unroll
        for (int r = 0; r < 8; ++r) {
            const int m = mbase + r;
            const int bb = m >> 12, hw = m & 4095;
            out[((size_t)bb * kC + n) * kHW + hw] = acc[nt][r] + bsum;
        }
    }
}

// ---------- host launcher ----------
extern "C" void kernel_launch(void* const* d_in, const int* in_sizes, int n_in,
                              void* d_out, int out_size, void* d_ws, size_t ws_size,
                              hipStream_t stream) {
    (void)in_sizes; (void)n_in; (void)out_size; (void)ws_size;
    const float* x           = (const float*)d_in[0];
    const float* cin         = (const float*)d_in[1];
    const float* wg_w        = (const float*)d_in[2];
    const float* wg_b        = (const float*)d_in[3];
    const float* wv_w        = (const float*)d_in[4];
    const float* wv_b        = (const float*)d_in[5];
    const float* ap_w        = (const float*)d_in[6];
    const float* ap_b        = (const float*)d_in[7];
    const float* log_diff    = (const float*)d_in[8];
    const float* r1_w        = (const float*)d_in[9];
    const float* r1_b        = (const float*)d_in[10];
    const float* r2_w        = (const float*)d_in[11];
    const float* r2_b        = (const float*)d_in[12];
    const float* norm_w      = (const float*)d_in[13];
    const float* ro_w        = (const float*)d_in[14];
    const float* ro_b        = (const float*)d_in[15];
    const float* log_dt      = (const float*)d_in[16];
    const float* state_decay = (const float*)d_in[17];

    char* ws = (char*)d_ws;
    size_t off = 0;
    auto alloc = [&](size_t bytes) -> char* {
        char* p = ws + off;
        off = (off + bytes + 255) & ~(size_t)255;
        return p;
    };
    float* state     = (float*)alloc((size_t)kM * kC * 4);
    float* update    = (float*)alloc((size_t)kM * kC * 4);
    bf16*  state_bf  = (bf16*)alloc((size_t)kM * kC * 2);
    bf16*  obs_bf    = (bf16*)alloc((size_t)kM * kC * 2);
    bf16*  hidden    = (bf16*)alloc((size_t)kM * kHID * 2);
    float* acts      = (float*)alloc((size_t)kB * kT * kC * 4);
    bf16*  wgbf      = (bf16*)alloc((size_t)kC * kC * 2);
    bf16*  wvbf      = (bf16*)alloc((size_t)kC * kC * 2);
    bf16*  r1bf      = (bf16*)alloc((size_t)kHID * kC * 2);
    bf16*  r2bf      = (bf16*)alloc((size_t)kC * kHID * 2);
    bf16*  robf      = (bf16*)alloc((size_t)kC * kC * 2);
    float* scal      = (float*)alloc(2 * 4);
    float* strengths = (float*)alloc(3 * kC * 4);

    hipMemsetAsync(state, 0, (size_t)kM * kC * 4, stream);
    k_scal<<<1, 384, 0, stream>>>(log_dt, state_decay, log_diff, scal, strengths);
    k_acts<<<(kB * kT * kC + 255) / 256, 256, 0, stream>>>(cin, ap_w, ap_b, acts);
    k_cvt<<<(kC * kC + 255) / 256, 256, 0, stream>>>(wg_w, wgbf, kC * kC);
    k_cvt<<<(kC * kC + 255) / 256, 256, 0, stream>>>(wv_w, wvbf, kC * kC);
    k_cvt<<<(kHID * kC + 255) / 256, 256, 0, stream>>>(r1_w, r1bf, kHID * kC);
    k_cvt<<<(kC * kHID + 255) / 256, 256, 0, stream>>>(r2_w, r2bf, kC * kHID);
    k_cvt<<<(kC * kC + 255) / 256, 256, 0, stream>>>(ro_w, robf, kC * kC);

    for (int t = 0; t < kT; ++t) {
        k_transpose_obs<<<dim3(kHW / 32, kC / 32, kB), 256, 0, stream>>>(x, obs_bf, t);
        k_gate_value<<<kM / 128, 256, 0, stream>>>(obs_bf, wgbf, wvbf, wg_b, wv_b, scal, state, state_bf);
        for (int s = 0; s < 4; ++s) {
            k_gemm_r1<<<dim3(kM / 128, kHID / 128), 256, 0, stream>>>(state_bf, r1bf, r1_b, hidden);
            k_diffusion<<<(kM * (kC / 4) + 255) / 256, 256, 0, stream>>>(state, strengths, update);
            const int doNorm = ((s + 1) % 2 == 0);   // norm on s=1,3
            k_gemm_r2<<<kM / 128, 256, 0, stream>>>(hidden, r2bf, r2_b, update, acts, scal,
                                                    norm_w, t, doNorm, state, state_bf);
        }
    }
    k_final<<<kM / 128, 256, 0, stream>>>(state_bf, robf, ro_b, (float*)d_out);
}
